// VoxtralCodec_3642132267290
// MI455X (gfx1250) — matrix operand order repaired and checked by reading, not compiled, not run on hardware
//
#include <hip/hip_runtime.h>
#include <hip/hip_bf16.h>
#include <stdint.h>

// ---------------------------------------------------------------------------
// VoxtralCodec forward on gfx1250 (CDNA5, wave32, WMMA)
//   out0: quantized float32 (16, 544, 4096)
//   out1: codes     int32   (16, 33, 4096)   (stored after out0 in d_out)
//
// K2 blocking: 64 rows (4 M-tiles) per block; each wave does 4 WMMAs per
// B-operand fetch -> 4x less L2 operand traffic than 1-tile blocking.
// ---------------------------------------------------------------------------

typedef __attribute__((ext_vector_type(16))) __bf16 v16bf;
typedef __attribute__((ext_vector_type(8)))  float  v8f;

#define BATCH   16
#define DFULL   544
#define DSEM    512
#define DACO    32
#define TLEN    4096
#define NCODES  4096
#define NTILESN 256        // 4096 / 16
#define KSTEPS  16         // 512 / 32
#define MTILES  4          // M-tiles per block (64 rows)
#define MROWS   64
#define HALFQ   7.5f
#define EPSU    1e-5f

static __device__ __forceinline__ unsigned short f32_to_bf16_rne(float f) {
    unsigned int u = __float_as_uint(f);
    u += 0x7FFFu + ((u >> 16) & 1u);   // round-to-nearest-even (ignore NaN edge)
    return (unsigned short)(u >> 16);
}

// ---------------------------------------------------------------------------
// K1: emb = embedding_sum / clip(usage, 1e-5);  per-row norm^2;
//     bf16 copy pre-swizzled into WMMA B-operand (32x16, 16-bit) layout:
//       lane = (code%16) + 16*h,  h = ((k%32)/8)%2,  elem e = k%8 + 8*((k%32)/16),
//       halfword index = ((nTile*16 + kStep)*32 + lane)*16 + e
// ---------------------------------------------------------------------------
__global__ __launch_bounds__(256)
void k1_normalize_emb(const float* __restrict__ emb_sum,
                      const float* __restrict__ usage,
                      float* __restrict__ emb_f32,
                      float* __restrict__ enorm,
                      unsigned short* __restrict__ emb_swz) {
    const int code = blockIdx.x;
    const int tid  = threadIdx.x;
    const float inv = 1.0f / fmaxf(usage[code], EPSU);
    const int nTile = code >> 4;
    const int n     = code & 15;

    float s = 0.0f;
    #pragma unroll
    for (int i = 0; i < 2; ++i) {
        const int k = tid + (i << 8);
        const float e = emb_sum[(size_t)code * DSEM + k] * inv;
        emb_f32[(size_t)code * DSEM + k] = e;
        s = fmaf(e, e, s);
        const int kStep = k >> 5;
        const int kk    = k & 31;
        const int h     = (kk >> 3) & 1;
        const int elem  = (kk & 7) + ((kk >> 4) << 3);
        const int lane  = n + (h << 4);
        emb_swz[((size_t)(nTile * KSTEPS + kStep) * 32 + lane) * 16 + elem] =
            f32_to_bf16_rne(e);
    }

    __shared__ float red[256];
    red[tid] = s;
    __syncthreads();
    for (int o = 128; o > 0; o >>= 1) {
        if (tid < o) red[tid] += red[tid + o];
        __syncthreads();
    }
    if (tid == 0) enorm[code] = red[0];
}

// ---------------------------------------------------------------------------
// K2: per 64-row M-block (64 consecutive t of one batch):
//   - stage A (64 x 512, bf16, WMMA A layout, 4 sub-tiles of 16 KB) in LDS
//   - 8 waves split the 256 N-tiles; per (nTile,kStep): 1 B fragment from L2,
//     4 A fragments from LDS, 4x v_wmma_f32_16x16x32_bf16 into 4 accumulators
//   - running argmin of enorm[n] - 2*dot, shfl_xor + LDS reductions -> codes
//   - gather winning emb rows via reused LDS buffer, coalesced writeback
// grid = 1024 blocks x 256 threads
// ---------------------------------------------------------------------------
__global__ __launch_bounds__(256)
void k2_sem_vq(const float* __restrict__ x,
               const float* __restrict__ emb_f32,
               const float* __restrict__ enorm,
               const unsigned short* __restrict__ emb_swz,
               const float* __restrict__ sem_probs,
               float* __restrict__ qOut,
               int* __restrict__ codesOut) {
    // 64 KB: A-operand tiles during GEMM, float gather staging afterwards
    __shared__ alignas(32) unsigned char ldsBuf[MTILES * KSTEPS * 512 * 2];
    __shared__ float redV[8][MROWS];
    __shared__ int   redI[8][MROWS];
    __shared__ int   finalCode[MROWS];

    unsigned short* ldsA = (unsigned short*)ldsBuf;

    const int tid  = threadIdx.x;
    const int lane = tid & 31;
    const int wid  = tid >> 5;                 // 8 waves
    const int b    = blockIdx.x >> 6;          // 64 blocks per batch
    const int t0   = (blockIdx.x & 63) << 6;   // 64 rows per block

    // ---- stage A (64 rows x 512 k) into WMMA A layout, 4 sub-tiles ----
    {
        const int tq    = tid & 63;             // row within block (t offset)
        const int kBase = tid >> 6;             // 4 k-lanes
        const int s     = tq >> 4;              // sub-tile
        const int m     = tq & 15;              // row within sub-tile
        const size_t xrow = ((size_t)b * DFULL) * TLEN + t0 + tq;
        #pragma unroll
        for (int i = 0; i < 128; ++i) {
            const int k = kBase + (i << 2);
            const float v = x[xrow + (size_t)k * TLEN];
            const unsigned short hv = f32_to_bf16_rne(v);
            // A operand 16x32 bf16 layout (ISA 7.12.2):
            const int kStep = k >> 5;
            const int kk    = k & 31;
            const int g     = kk >> 4;          // VGPR group
            const int r16   = kk & 15;
            const int h     = r16 >> 3;         // lane half
            const int vv    = (r16 & 7) >> 1;   // VGPR within group
            const int j     = kk & 1;           // low/high half of dword
            const int alane = m + (h << 4);
            const int e     = (((g << 2) + vv) << 1) + j;
            ldsA[s * (KSTEPS * 512) + kStep * 512 + alane * 16 + e] = hv;
        }
    }
    __syncthreads();

    // ---- WMMA sweep: 4 M-tiles per B fragment ----
    float bestv[MTILES][8];
    int   besti[MTILES][8];
    #pragma unroll
    for (int s = 0; s < MTILES; ++s)
        #pragma unroll
        for (int r = 0; r < 8; ++r) { bestv[s][r] = 3.4e38f; besti[s][r] = 0; }

    for (int nt = wid; nt < NTILESN; nt += 8) {
        const unsigned short* bp = emb_swz + (size_t)nt * (KSTEPS * 512) + lane * 16;
        const float en = enorm[(nt << 4) + (lane & 15)];
        v8f c0 = {}, c1 = {}, c2 = {}, c3 = {};
        #pragma unroll 2   // keep A fragments loop-carried (no 512-reg hoist)
        for (int ks = 0; ks < KSTEPS; ++ks) {
            const v16bf bb = *reinterpret_cast<const v16bf*>(bp + ks * 512);
            const unsigned short* ap = ldsA + ks * 512 + lane * 16;
            const v16bf a0 = *reinterpret_cast<const v16bf*>(ap);
            const v16bf a1 = *reinterpret_cast<const v16bf*>(ap + 1 * KSTEPS * 512);
            const v16bf a2 = *reinterpret_cast<const v16bf*>(ap + 2 * KSTEPS * 512);
            const v16bf a3 = *reinterpret_cast<const v16bf*>(ap + 3 * KSTEPS * 512);
            c0 = __builtin_amdgcn_wmma_f32_16x16x32_bf16(false, a0, false, bb, (short)0, c0, false, false);
            c1 = __builtin_amdgcn_wmma_f32_16x16x32_bf16(false, a1, false, bb, (short)0, c1, false, false);
            c2 = __builtin_amdgcn_wmma_f32_16x16x32_bf16(false, a2, false, bb, (short)0, c2, false, false);
            c3 = __builtin_amdgcn_wmma_f32_16x16x32_bf16(false, a3, false, bb, (short)0, c3, false, false);
        }
        const int nIdx = (nt << 4) + (lane & 15);
        #pragma unroll
        for (int r = 0; r < 8; ++r) {
            float m2;
            m2 = fmaf(-2.0f, c0[r], en); if (m2 < bestv[0][r]) { bestv[0][r] = m2; besti[0][r] = nIdx; }
            m2 = fmaf(-2.0f, c1[r], en); if (m2 < bestv[1][r]) { bestv[1][r] = m2; besti[1][r] = nIdx; }
            m2 = fmaf(-2.0f, c2[r], en); if (m2 < bestv[2][r]) { bestv[2][r] = m2; besti[2][r] = nIdx; }
            m2 = fmaf(-2.0f, c3[r], en); if (m2 < bestv[3][r]) { bestv[3][r] = m2; besti[3][r] = nIdx; }
        }
    }

    // ---- reduce across the 16 lanes sharing each row M (xor 1,2,4,8) ----
    #pragma unroll
    for (int off = 1; off <= 8; off <<= 1) {
        #pragma unroll
        for (int s = 0; s < MTILES; ++s)
            #pragma unroll
            for (int r = 0; r < 8; ++r) {
                const float ov = __shfl_xor(bestv[s][r], off, 32);
                const int   oi = __shfl_xor(besti[s][r], off, 32);
                if (ov < bestv[s][r] || (ov == bestv[s][r] && oi < besti[s][r])) {
                    bestv[s][r] = ov; besti[s][r] = oi;
                }
            }
    }
    // lanes 0 / 16 hold rows M = s*16 + r / s*16 + 8 + r  (C-matrix layout)
    if ((lane & 15) == 0) {
        const int hbase = (lane >> 4) * 8;
        #pragma unroll
        for (int s = 0; s < MTILES; ++s)
            #pragma unroll
            for (int r = 0; r < 8; ++r) {
                redV[wid][s * 16 + hbase + r] = bestv[s][r];
                redI[wid][s * 16 + hbase + r] = besti[s][r];
            }
    }
    __syncthreads();

    // ---- cross-wave reduction + code write ----
    if (tid < MROWS) {
        float bv = 3.4e38f; int bi = 0;
        #pragma unroll
        for (int w = 0; w < 8; ++w) {
            const float v = redV[w][tid];
            const int   i = redI[w][tid];
            if (v < bv || (v == bv && i < bi)) { bv = v; bi = i; }
        }
        finalCode[tid] = bi;
        codesOut[((size_t)b * 33) * TLEN + t0 + tid] = bi;   // channel 0 = sem codes
    }

    const bool useQ = sem_probs[b] < 0.5f;   // uniform across block
    float* ldsQ = (float*)ldsBuf;            // reuse A buffer: 16 x 516 floats/chunk

    // ---- gather + coalesced writeback in 4 chunks of 16 rows ----
    #pragma unroll 1
    for (int cc = 0; cc < 4; ++cc) {
        __syncthreads();
        if (useQ) {
            const int mRow = tid >> 4;
            const int kl   = tid & 15;
            const size_t ebase = (size_t)finalCode[cc * 16 + mRow] * DSEM;
            #pragma unroll
            for (int i = 0; i < 32; ++i) {
                const int k = kl + (i << 4);
                ldsQ[mRow * 516 + k] = emb_f32[ebase + k];
            }
        }
        __syncthreads();
        {
            const int tloc  = tid & 15;
            const int kBase = tid >> 4;
            const size_t obase = ((size_t)b * DFULL) * TLEN + t0 + cc * 16 + tloc;
            #pragma unroll
            for (int i = 0; i < 32; ++i) {
                const int k = kBase + (i << 4);
                const size_t o = obase + (size_t)k * TLEN;
                qOut[o] = useQ ? ldsQ[tloc * 516 + k] : x[o];
            }
        }
    }
}

// ---------------------------------------------------------------------------
// K3: acoustic FSQ path, elementwise over 16 x 32 x 4096
// ---------------------------------------------------------------------------
__global__ __launch_bounds__(256)
void k3_aco_fsq(const float* __restrict__ x,
                const float* __restrict__ aco_probs,
                const float* __restrict__ noise,
                float* __restrict__ qOut,
                int* __restrict__ codesOut) {
    const int idx = blockIdx.x * blockDim.x + threadIdx.x;
    if (idx >= BATCH * DACO * TLEN) return;
    const int t = idx & (TLEN - 1);
    const int c = (idx >> 12) & 31;
    const int b = idx >> 17;

    const float p  = aco_probs[b];
    const float xv = x[((size_t)b * DFULL + DSEM + c) * TLEN + t];
    const float zb = tanhf(xv) * HALFQ;

    float zout;
    if (p < 0.5f) {
        zout = rintf(zb);                              // jnp.round (RNE)
    } else if (p < 0.75f) {
        const float nv = noise[((size_t)b * DACO + c) * TLEN + t];
        zout = fminf(fmaxf(zb + nv, -HALFQ), HALFQ);
    } else {
        zout = zb;
    }
    const int code = (int)fminf(fmaxf(rintf(zout + HALFQ), 0.0f), 15.0f);

    qOut[((size_t)b * DFULL + DSEM + c) * TLEN + t]  = zout / HALFQ;
    codesOut[((size_t)b * 33 + 1 + c) * TLEN + t]    = code;
}

// ---------------------------------------------------------------------------
extern "C" void kernel_launch(void* const* d_in, const int* in_sizes, int n_in,
                              void* d_out, int out_size, void* d_ws, size_t ws_size,
                              hipStream_t stream) {
    (void)in_sizes; (void)n_in; (void)out_size; (void)ws_size;

    const float* x         = (const float*)d_in[0];
    const float* emb_sum   = (const float*)d_in[1];
    const float* usage     = (const float*)d_in[2];
    const float* sem_probs = (const float*)d_in[3];
    const float* aco_probs = (const float*)d_in[4];
    const float* noise     = (const float*)d_in[5];

    float* qOut    = (float*)d_out;                               // (16,544,4096) f32
    int*   codes   = (int*)(qOut + (size_t)BATCH * DFULL * TLEN); // (16,33,4096) i32

    // workspace: emb_f32 (8 MB) | enorm (16 KB) | emb_swz bf16 (4 MB)
    char* ws = (char*)d_ws;
    float*          emb_f32 = (float*)ws;
    float*          enorm   = (float*)(ws + (size_t)NCODES * DSEM * 4);
    unsigned short* emb_swz = (unsigned short*)(ws + (size_t)NCODES * DSEM * 4
                                                   + (size_t)NCODES * 4);

    k1_normalize_emb<<<NCODES, 256, 0, stream>>>(emb_sum, usage,
                                                 emb_f32, enorm, emb_swz);
    k2_sem_vq<<<(BATCH * TLEN) / MROWS, 256, 0, stream>>>(x, emb_f32, enorm, emb_swz,
                                                          sem_probs, qOut, codes);
    k3_aco_fsq<<<(BATCH * DACO * TLEN) / 256, 256, 0, stream>>>(x, aco_probs, noise,
                                                                qOut, codes);
}
